// TEDotProductAttention_64261300683100
// MI455X (gfx1250) — matrix operand order repaired and checked by reading, not compiled, not run on hardware
//
#include <hip/hip_runtime.h>

typedef __attribute__((ext_vector_type(16))) _Float16 v16h;
typedef __attribute__((ext_vector_type(8)))  _Float16 v8h;
typedef __attribute__((ext_vector_type(4)))  _Float16 v4h;
typedef __attribute__((ext_vector_type(8)))  float    v8f;

// Problem constants: S=2048, B=2, H=16, D=64
#define SEQ 2048
#define BAT 2
#define NHD 16
#define HD  64
#define RS  (BAT*NHD*HD)     // element stride between consecutive sequence rows
#define KT  64               // key tile
#define NT  (SEQ/KT)         // 32 tiles
#define STR 72               // padded LDS stride (halves): 144B -> conflict-free b128 reads

#define SC2  0.18033688011112042f   /* (1/8) * log2(e) */
#define NEG2 (-1.442695e9f)         /* -1e9 * log2(e)  */

#if __has_builtin(__builtin_amdgcn_exp2f)
#define EXP2F(x) __builtin_amdgcn_exp2f(x)
#else
#define EXP2F(x) exp2f(x)
#endif

// ---- xor-shuffle within each 16-lane half (matches WMMA C-layout rows) ----
#if __has_builtin(__builtin_amdgcn_permlane16)
__device__ __forceinline__ float xs16(float x, unsigned slo, unsigned shi) {
    unsigned i = (unsigned)__float_as_int(x);
    i = __builtin_amdgcn_permlane16(i, i, slo, shi, false, false);
    return __int_as_float((int)i);
}
#define XS1(x) xs16((x), 0x67452301u, 0xEFCDAB89u)
#define XS2(x) xs16((x), 0x54761032u, 0xDCFE98BAu)
#define XS4(x) xs16((x), 0x32107654u, 0xBA98FEDCu)
#define XS8(x) xs16((x), 0xFEDCBA98u, 0x76543210u)
#else
#define XS1(x) __shfl_xor((x), 1, 32)
#define XS2(x) __shfl_xor((x), 2, 32)
#define XS4(x) __shfl_xor((x), 4, 32)
#define XS8(x) __shfl_xor((x), 8, 32)
#endif

__device__ __forceinline__ float rmax16(float x) {
    x = fmaxf(x, XS1(x)); x = fmaxf(x, XS2(x));
    x = fmaxf(x, XS4(x)); x = fmaxf(x, XS8(x));
    return x;
}
__device__ __forceinline__ float rsum16(float x) {
    x += XS1(x); x += XS2(x); x += XS4(x); x += XS8(x);
    return x;
}

__device__ __forceinline__ unsigned pk_f16(float a, float b) {
#if __has_builtin(__builtin_amdgcn_cvt_pkrtz)
    return __builtin_bit_cast(unsigned, __builtin_amdgcn_cvt_pkrtz(a, b));
#else
    v4h tmp; tmp[0] = (_Float16)a; tmp[1] = (_Float16)b; tmp[2] = 0; tmp[3] = 0;
    return __builtin_bit_cast(unsigned long long, tmp) & 0xFFFFFFFFu;
#endif
}

__global__ __launch_bounds__(128)
void TEDotProductAttention_fa_kernel(const float* __restrict__ q,
                                     const float* __restrict__ k,
                                     const float* __restrict__ v,
                                     const unsigned char* __restrict__ mask,
                                     float* __restrict__ out)
{
    __shared__ __align__(16) _Float16 sK [KT * STR];     // [key][d] row-major
    __shared__ __align__(16) _Float16 sVt[HD * STR];     // [d][key] transposed
    __shared__ __align__(16) _Float16 sP [4][16 * STR];  // per-wave P tile [q][key]

    const int tid  = threadIdx.x;
    const int wave = tid >> 5;
    const int lane = tid & 31;
    const int ln   = lane & 15;
    const int hi   = lane >> 4;

    const int bid = blockIdx.x;
    const int qb  = bid / (BAT * NHD);
    const int rem = bid % (BAT * NHD);
    const int b   = rem / NHD;
    const int h   = rem % NHD;

    const int q0 = qb * 64 + wave * 16;

    const float* kbase = k + (size_t)(b * NHD + h) * HD;
    const float* vbase = v + (size_t)(b * NHD + h) * HD;
    const unsigned char* mb = mask + (size_t)b * SEQ * SEQ;

    // ---- Q fragments: A-layout 16x32 f16, two K-chunks covering D=64 ----
    v16h qA[2];
    {
        const float* qrow = q + ((size_t)((q0 + ln) * BAT + b) * NHD + h) * HD;
        #pragma unroll
        for (int dc = 0; dc < 2; ++dc) {
            #pragma unroll
            for (int g = 0; g < 2; ++g) {
                const int d0 = dc * 32 + g * 16 + 8 * hi;
                float4 f0 = *(const float4*)(qrow + d0);
                float4 f1 = *(const float4*)(qrow + d0 + 4);
                qA[dc][g*8+0] = (_Float16)f0.x; qA[dc][g*8+1] = (_Float16)f0.y;
                qA[dc][g*8+2] = (_Float16)f0.z; qA[dc][g*8+3] = (_Float16)f0.w;
                qA[dc][g*8+4] = (_Float16)f1.x; qA[dc][g*8+5] = (_Float16)f1.y;
                qA[dc][g*8+6] = (_Float16)f1.z; qA[dc][g*8+7] = (_Float16)f1.w;
            }
        }
    }

    v8f o[4];
    #pragma unroll
    for (int i = 0; i < 4; ++i) o[i] = (v8f){0.f,0.f,0.f,0.f,0.f,0.f,0.f,0.f};
    float mrow[8], lrow[8];
    #pragma unroll
    for (int r = 0; r < 8; ++r) { mrow[r] = -1e30f; lrow[r] = 0.0f; }

    for (int t = 0; t < NT; ++t) {
        const int n0 = t * KT;

        // ---- mask fast-path probe: this wave's 16 rows x 64 cols as dwords ----
        bool okl = true;
        #pragma unroll
        for (int rr = 0; rr < 8; ++rr) {
            const int row = q0 + rr * 2 + hi;
            unsigned int w = *(const unsigned int*)(mb + (size_t)row * SEQ + n0 + 4 * ln);
            okl = okl && (w == 0x01010101u);
        }
        const int allok = __all(okl);

        // ---- cooperative stage: K row-major (packed b64 stores), V transposed ----
        #pragma unroll
        for (int c = 0; c < 8; ++c) {
            const int e  = (tid + 128 * c) * 4;
            const int kr = e >> 6;
            const int dc = e & 63;
            const float* kp = kbase + (size_t)(n0 + kr) * RS + dc;
            const float* vp = vbase + (size_t)(n0 + kr) * RS + dc;
            float4 kf = *(const float4*)kp;
            float4 vf = *(const float4*)vp;
            uint2 kpk;
            kpk.x = pk_f16(kf.x, kf.y);
            kpk.y = pk_f16(kf.z, kf.w);
            *(uint2*)(&sK[kr * STR + dc]) = kpk;
            sVt[(dc + 0) * STR + kr] = (_Float16)vf.x;
            sVt[(dc + 1) * STR + kr] = (_Float16)vf.y;
            sVt[(dc + 2) * STR + kr] = (_Float16)vf.z;
            sVt[(dc + 3) * STR + kr] = (_Float16)vf.w;
            if ((t + 1 < NT) && ((dc & 31) == 0)) {   // one prefetch per 128B line
                __builtin_prefetch(kp + (size_t)KT * RS, 0, 3);
                __builtin_prefetch(vp + (size_t)KT * RS, 0, 3);
            }
        }
        __syncthreads();

        // ---- scores S[16x64] = Q . K^T : 8 wmma f32_16x16x32_f16 ----
        v8f s[4];
        #pragma unroll
        for (int nh = 0; nh < 4; ++nh) {
            v8f acc = (v8f){0.f,0.f,0.f,0.f,0.f,0.f,0.f,0.f};
            #pragma unroll
            for (int dc2 = 0; dc2 < 2; ++dc2) {
                const _Float16* p = &sK[(nh * 16 + ln) * STR + dc2 * 32 + 8 * hi];
                v16h bK;
                ((v8h*)&bK)[0] = *(const v8h*)(p);
                ((v8h*)&bK)[1] = *(const v8h*)(p + 16);
                acc = __builtin_amdgcn_wmma_f32_16x16x32_f16(
                        false, qA[dc2], false, bK, (short)0, acc, false, false);
            }
            s[nh] = acc;
        }

        // ---- online softmax in log2 domain ----
        #pragma unroll
        for (int r = 0; r < 8; ++r) {
            float sc[4];
            #pragma unroll
            for (int j = 0; j < 4; ++j) sc[j] = s[j][r] * SC2;
            if (!allok) {   // wave-uniform slow path for non-trivial masks
                const unsigned char* mrp = mb + (size_t)(q0 + r + 8 * hi) * SEQ + n0 + ln;
                #pragma unroll
                for (int j = 0; j < 4; ++j)
                    if (!mrp[16 * j]) sc[j] = NEG2;
            }
            float tm = fmaxf(fmaxf(sc[0], sc[1]), fmaxf(sc[2], sc[3]));
            tm = rmax16(tm);
            const float mn = fmaxf(mrow[r], tm);
            const float al = EXP2F(mrow[r] - mn);
            const float p0 = EXP2F(sc[0] - mn);
            const float p1 = EXP2F(sc[1] - mn);
            const float p2 = EXP2F(sc[2] - mn);
            const float p3 = EXP2F(sc[3] - mn);
            float rs = (p0 + p1) + (p2 + p3);
            rs = rsum16(rs);
            lrow[r] = lrow[r] * al + rs;
            mrow[r] = mn;
            #pragma unroll
            for (int db = 0; db < 4; ++db) o[db][r] *= al;
            _Float16* pw = &sP[wave][(r + 8 * hi) * STR];
            pw[ln]      = (_Float16)p0;
            pw[ln + 16] = (_Float16)p1;
            pw[ln + 32] = (_Float16)p2;
            pw[ln + 48] = (_Float16)p3;
        }
        __syncthreads();

        // ---- P(16x64) . V(64x64) : 8 wmma ----
        v16h pA[2];
        #pragma unroll
        for (int kc = 0; kc < 2; ++kc) {
            const _Float16* pp = &sP[wave][ln * STR + kc * 32 + 8 * hi];
            ((v8h*)&pA[kc])[0] = *(const v8h*)(pp);
            ((v8h*)&pA[kc])[1] = *(const v8h*)(pp + 16);
        }
        #pragma unroll
        for (int db = 0; db < 4; ++db) {
            #pragma unroll
            for (int kc = 0; kc < 2; ++kc) {
                const _Float16* vp = &sVt[(db * 16 + ln) * STR + kc * 32 + 8 * hi];
                v16h bV;
                ((v8h*)&bV)[0] = *(const v8h*)(vp);
                ((v8h*)&bV)[1] = *(const v8h*)(vp + 16);
                o[db] = __builtin_amdgcn_wmma_f32_16x16x32_f16(
                            false, pA[kc], false, bV, (short)0, o[db], false, false);
            }
        }
        __syncthreads();
    }

    // ---- normalize + store fp32 output [S, B, H*D] ----
    #pragma unroll
    for (int r = 0; r < 8; ++r) {
        const int row = q0 + r + 8 * hi;
        const float inv = 1.0f / lrow[r];
        float* op = out + ((size_t)row * BAT + b) * (NHD * HD) + h * HD + ln;
        #pragma unroll
        for (int db = 0; db < 4; ++db)
            op[db * 16] = o[db][r] * inv;
    }
}

extern "C" void kernel_launch(void* const* d_in, const int* in_sizes, int n_in,
                              void* d_out, int out_size, void* d_ws, size_t ws_size,
                              hipStream_t stream) {
    (void)in_sizes; (void)n_in; (void)out_size; (void)d_ws; (void)ws_size;
    const float* q = (const float*)d_in[0];
    const float* k = (const float*)d_in[1];
    const float* v = (const float*)d_in[2];
    const unsigned char* mask = (const unsigned char*)d_in[3];
    float* out = (float*)d_out;

    dim3 grid((SEQ / 64) * BAT * NHD);   // 1024 workgroups: (qblock, b, h)
    dim3 block(128);                     // 4 waves of 32
    TEDotProductAttention_fa_kernel<<<grid, block, 0, stream>>>(q, k, v, mask, out);
}
